// Attention_73229192397093
// MI455X (gfx1250) — matrix operand order repaired and checked by reading, not compiled, not run on hardware
//
#include <hip/hip_runtime.h>
#include <hip/hip_bf16.h>

// Problem constants (match reference)
#define NB 4
#define NS 2048
#define NH 16
#define ND 64
#define NDM 1024

typedef __bf16 bf16;
typedef __attribute__((ext_vector_type(16))) __bf16 v16bf;
typedef __attribute__((ext_vector_type(8)))  __bf16 v8bf;
typedef __attribute__((ext_vector_type(8)))  float   v8f;

static __device__ __forceinline__ v8f zero_v8f() {
  v8f z = {0.f, 0.f, 0.f, 0.f, 0.f, 0.f, 0.f, 0.f};
  return z;
}

static __device__ __forceinline__ v8f wmma_bf16(v16bf a, v16bf b, v8f c) {
  // D = A(16x32 bf16) x B(32x16 bf16) + C(16x16 f32)
  return __builtin_amdgcn_wmma_f32_16x16x32_bf16(false, a, false, b, (short)0, c,
                                                 false, false);
}

// A-fragment (16x32, bf16) from row-major memory: lane l holds row m=l&15,
// elements i -> k = kbase + (l>>4)*16 + i. Two 16B loads.
static __device__ __forceinline__ v16bf load_a_bf16(const bf16* rowptr, int g, int kbase) {
  const v8bf lo = *(const v8bf*)(rowptr + kbase + g * 16);
  const v8bf hi = *(const v8bf*)(rowptr + kbase + g * 16 + 8);
  v16bf r;
#pragma unroll
  for (int i = 0; i < 8; ++i) { r[i] = lo[i]; r[i + 8] = hi[i]; }
  return r;
}

// Same A-fragment but sourced from f32 memory (converted to bf16).
static __device__ __forceinline__ v16bf load_a_f32(const float* rowptr, int g, int kbase) {
  const float* p = rowptr + kbase + g * 16;
  v16bf r;
#pragma unroll
  for (int i = 0; i < 8; ++i) { r[i] = (__bf16)p[i]; r[i + 8] = (__bf16)p[i + 8]; }
  return r;
}

// B-fragment slice: 16 contiguous f32 -> bf16 (lane holds col n=l&15, k = base + i)
static __device__ __forceinline__ v16bf cvt_b16_from_f32(const float* p) {
  v16bf r;
#pragma unroll
  for (int i = 0; i < 16; ++i) r[i] = (__bf16)p[i];
  return r;
}

// ---------------------------------------------------------------------------
// Kernel 0: Wo (f32, [1024,1024] row-major = [n][k] of Wo^T B-frags) -> bf16
// ---------------------------------------------------------------------------
__global__ void wo_to_bf16_kernel(const float* __restrict__ Wo,
                                  bf16* __restrict__ Wob, int n) {
  int i = blockIdx.x * blockDim.x + threadIdx.x;
  if (i < n) Wob[i] = (__bf16)Wo[i];
}

// ---------------------------------------------------------------------------
// Kernel 1: per-head Q/K/V projections via WMMA.
//   wave = one (b,h) and 16 rows of S; computes X(16x64) @ W^T(64x64) + bias
//   Qp,Kp layout [B,H,S,D] bf16 ; Vt layout [B,H,D,S] bf16 (for V^T A-frags)
// ---------------------------------------------------------------------------
__global__ void __launch_bounds__(128) qkv_proj_kernel(
    const float* __restrict__ q_in, const float* __restrict__ k_in,
    const float* __restrict__ v_in,
    const float* __restrict__ Wq, const float* __restrict__ bq,
    const float* __restrict__ Wk, const float* __restrict__ bk,
    const float* __restrict__ Wv, const float* __restrict__ bv,
    bf16* __restrict__ Qp, bf16* __restrict__ Kp, bf16* __restrict__ Vt) {
  const int lane = threadIdx.x & 31;
  const int wave = threadIdx.x >> 5;
  const int n = lane & 15;      // C/D column (dout)  | A row (s) | B column
  const int g = lane >> 4;      // half-wave select
  const int blk = blockIdx.x;   // NB*NH*(NS/64)
  const int sblk = blk % (NS / 64);
  const int bh = blk / (NS / 64);
  const int b = bh / NH;
  const int h = bh % NH;
  const int sbase = sblk * 64 + wave * 16;

  const float* X[3]  = { q_in, k_in, v_in };
  const float* W[3]  = { Wq, Wk, Wv };
  const float* BI[3] = { bq, bk, bv };

#pragma unroll
  for (int mat = 0; mat < 3; ++mat) {
    // A = input rows (s), K-dim = d_in (64 -> 2 chunks of 32)
    const float* xrow = X[mat] + ((size_t)b * NS + sbase + n) * NDM + h * ND;
    const v16bf a0 = load_a_f32(xrow, g, 0);
    const v16bf a1 = load_a_f32(xrow, g, 32);
#pragma unroll
    for (int nt = 0; nt < 4; ++nt) {
      // B = W^T : B[k=din][n=dout] = W[dout][din] -> contiguous in din
      const int dout = nt * 16 + n;
      const float* wrow = W[mat] + dout * ND;
      const v16bf b0 = cvt_b16_from_f32(wrow + g * 16);
      const v16bf b1 = cvt_b16_from_f32(wrow + 32 + g * 16);
      v8f acc = zero_v8f();
      acc = wmma_bf16(a0, b0, acc);
      acc = wmma_bf16(a1, b1, acc);
      const float bias = BI[mat][dout];
      if (mat < 2) {
        bf16* dst = (mat == 0 ? Qp : Kp) + (size_t)bh * NS * ND;
#pragma unroll
        for (int r = 0; r < 8; ++r)
          dst[(size_t)(sbase + r + 8 * g) * ND + dout] = (__bf16)(acc[r] + bias);
      } else {
        // Vt [B,H,D,S]: rows r are contiguous in s -> single 16B store
        v8bf pk;
#pragma unroll
        for (int r = 0; r < 8; ++r) pk[r] = (__bf16)(acc[r] + bias);
        *(v8bf*)(Vt + ((size_t)bh * ND + dout) * NS + sbase + 8 * g) = pk;
      }
    }
  }
}

// ---------------------------------------------------------------------------
// Kernel 2: flash attention. wave = 16 queries of one (b,h).
//   Computes S^T = K x Q^T so softmax-over-keys is an in-lane reduction,
//   then O^T += V^T x P^T with P^T built by one half-wave exchange.
//   Scale = 1/(D/2) = 1/32 (faithful); mask intentionally ignored (faithful).
// ---------------------------------------------------------------------------
__global__ void __launch_bounds__(128) flash_attn_kernel(
    const bf16* __restrict__ Qp, const bf16* __restrict__ Kp,
    const bf16* __restrict__ Vt, bf16* __restrict__ attn) {
  const int lane = threadIdx.x & 31;
  const int wave = threadIdx.x >> 5;
  const int n = lane & 15;      // query index within tile (C/D column)
  const int g = lane >> 4;
  const int blk = blockIdx.x;   // NB*NH*(NS/64)
  const int qblk = blk % (NS / 64);
  const int bh = blk / (NS / 64);
  const int b = bh / NH;
  const int h = bh % NH;
  const int qbase = qblk * 64 + wave * 16;

  const bf16* Qb = Qp + (size_t)bh * NS * ND;
  const bf16* Kb = Kp + (size_t)bh * NS * ND;
  const bf16* Vb = Vt + (size_t)bh * ND * NS;

  // Q^T B-fragments (reused for all key blocks): lane holds query row n,
  // contiguous d. qb[c]: k=d = c*32 + g*16 + i
  const bf16* qrow = Qb + (size_t)(qbase + n) * ND;
  const v16bf qb0 = *(const v16bf*)(qrow + g * 16);
  const v16bf qb1 = *(const v16bf*)(qrow + 32 + g * 16);

  v8f o0 = zero_v8f(), o1 = zero_v8f(), o2 = zero_v8f(), o3 = zero_v8f();
  float m_i = -3.0e38f;
  float l_i = 0.0f;
  const float scale = 1.0f / 32.0f;   // score / (D/2), NOT 1/sqrt(D)

  for (int kb = 0; kb < NS / 32; ++kb) {
    const int key0 = kb * 32;
    if (kb + 1 < NS / 32)   // gfx1250 global_prefetch_b8 on next K tile
      __builtin_prefetch(Kb + (size_t)(key0 + 32 + n) * ND, 0, 0);

    // S^T tiles: rows = keys, cols = queries. st[t] covers keys [16t,16t+16)
    const bf16* krow0 = Kb + (size_t)(key0 + n) * ND;
    v8f st0 = zero_v8f();
    st0 = wmma_bf16(load_a_bf16(krow0, g, 0),  qb0, st0);
    st0 = wmma_bf16(load_a_bf16(krow0, g, 32), qb1, st0);
    const bf16* krow1 = Kb + (size_t)(key0 + 16 + n) * ND;
    v8f st1 = zero_v8f();
    st1 = wmma_bf16(load_a_bf16(krow1, g, 0),  qb0, st1);
    st1 = wmma_bf16(load_a_bf16(krow1, g, 32), qb1, st1);

    // Online softmax over keys: 16 in-lane values + one cross-half exchange
    float p0[8], p1[8];
    float mx = m_i;
#pragma unroll
    for (int r = 0; r < 8; ++r) {
      p0[r] = st0[r] * scale;
      p1[r] = st1[r] * scale;
      mx = fmaxf(mx, fmaxf(p0[r], p1[r]));
    }
    mx = fmaxf(mx, __shfl_xor(mx, 16, 32));
    float rs = 0.f;
#pragma unroll
    for (int r = 0; r < 8; ++r) {
      p0[r] = __expf(p0[r] - mx);
      p1[r] = __expf(p1[r] - mx);
      rs += p0[r] + p1[r];
    }
    rs += __shfl_xor(rs, 16, 32);
    const float alpha = __expf(m_i - mx);
    l_i = l_i * alpha + rs;
    m_i = mx;
#pragma unroll
    for (int r = 0; r < 8; ++r) {
      o0[r] *= alpha; o1[r] *= alpha; o2[r] *= alpha; o3[r] *= alpha;
    }

    // Build P^T B-fragment (32 keys x 16 queries). B wants lane group g to
    // hold keys [16g,16g+16): own tile regs for one half, partner lane
    // (l^16) regs for the other -> single xor-16 exchange.
    v16bf pb;
#pragma unroll
    for (int r = 0; r < 8; ++r) {
      const float x0 = __shfl_xor(p0[r], 16, 32);
      const float x1 = __shfl_xor(p1[r], 16, 32);
      pb[r]     = (__bf16)(g == 0 ? p0[r] : x1);
      pb[r + 8] = (__bf16)(g == 0 ? x0 : p1[r]);
    }

    // O^T(16d x 16q) += V^T(16d x 32k) x P^T(32k x 16q), 4 d-tiles
    o0 = wmma_bf16(load_a_bf16(Vb + (size_t)(0 * 16 + n) * NS + key0, g, 0), pb, o0);
    o1 = wmma_bf16(load_a_bf16(Vb + (size_t)(1 * 16 + n) * NS + key0, g, 0), pb, o1);
    o2 = wmma_bf16(load_a_bf16(Vb + (size_t)(2 * 16 + n) * NS + key0, g, 0), pb, o2);
    o3 = wmma_bf16(load_a_bf16(Vb + (size_t)(3 * 16 + n) * NS + key0, g, 0), pb, o3);
  }

  // Epilogue: O /= l, write bf16 to attn [B,S,H*D]; rows r contiguous in d.
  const float inv = 1.0f / l_i;
  bf16* arow = attn + (size_t)(b * NS + qbase + n) * NDM + h * ND + g * 8;
  v8f oacc[4] = { o0, o1, o2, o3 };
#pragma unroll
  for (int td = 0; td < 4; ++td) {
    v8bf pk;
#pragma unroll
    for (int r = 0; r < 8; ++r) pk[r] = (__bf16)(oacc[td][r] * inv);
    *(v8bf*)(arow + td * 16) = pk;
  }
}

// ---------------------------------------------------------------------------
// Kernel 3: output projection Y[8192,1024] = X(bf16) @ Wo^T(bf16) + bo, f32 out
//   wave = 16 rows x 64 cols, K looped in chunks of 32.
// ---------------------------------------------------------------------------
__global__ void __launch_bounds__(128) out_proj_kernel(
    const bf16* __restrict__ Xb, const bf16* __restrict__ Wob,
    const float* __restrict__ bo, float* __restrict__ out) {
  const int lane = threadIdx.x & 31;
  const int wave = threadIdx.x >> 5;
  const int n = lane & 15;
  const int g = lane >> 4;
  const int blk = blockIdx.x;       // (M/16) * (N/256) = 512 * 4
  const int mtile = blk >> 2;
  const int nblk = blk & 3;
  const int mbase = mtile * 16;
  const int nbase = nblk * 256 + wave * 64;

  const bf16* xrow = Xb + (size_t)(mbase + n) * NDM;
  v8f acc[4] = { zero_v8f(), zero_v8f(), zero_v8f(), zero_v8f() };

  for (int kc = 0; kc < NDM / 32; ++kc) {
    const v16bf a = load_a_bf16(xrow, g, kc * 32);
    const int koff = kc * 32 + g * 16;
#pragma unroll
    for (int nt = 0; nt < 4; ++nt) {
      const v16bf bfrag =
          *(const v16bf*)(Wob + (size_t)(nbase + nt * 16 + n) * NDM + koff);
      acc[nt] = wmma_bf16(a, bfrag, acc[nt]);
    }
  }
#pragma unroll
  for (int nt = 0; nt < 4; ++nt) {
    const int col = nbase + nt * 16 + n;
    const float bias = bo[col];
#pragma unroll
    for (int r = 0; r < 8; ++r)
      out[(size_t)(mbase + r + 8 * g) * NDM + col] = acc[nt][r] + bias;
  }
}

// ---------------------------------------------------------------------------
extern "C" void kernel_launch(void* const* d_in, const int* in_sizes, int n_in,
                              void* d_out, int out_size, void* d_ws, size_t ws_size,
                              hipStream_t stream) {
  (void)in_sizes; (void)n_in; (void)out_size; (void)ws_size;
  const float* q_in = (const float*)d_in[0];
  const float* k_in = (const float*)d_in[1];
  const float* v_in = (const float*)d_in[2];
  // d_in[3] = mask: intentionally unused (reference discards masked_fill result)
  const float* Wq = (const float*)d_in[4];
  const float* bq = (const float*)d_in[5];
  const float* Wk = (const float*)d_in[6];
  const float* bk = (const float*)d_in[7];
  const float* Wv = (const float*)d_in[8];
  const float* bv = (const float*)d_in[9];
  const float* Wo = (const float*)d_in[10];
  const float* bo = (const float*)d_in[11];
  float* out = (float*)d_out;

  // Workspace layout (bytes): Qp 16M | Kp 16M | Vt 16M | attn 16M | Wob 2M
  char* ws = (char*)d_ws;
  const size_t seg = (size_t)NB * NH * NS * ND * sizeof(bf16); // 16 MiB
  bf16* Qp  = (bf16*)(ws + 0 * seg);
  bf16* Kp  = (bf16*)(ws + 1 * seg);
  bf16* Vt  = (bf16*)(ws + 2 * seg);
  bf16* At  = (bf16*)(ws + 3 * seg);     // NB*NS*NDM bf16 == same size
  bf16* Wob = (bf16*)(ws + 4 * seg);

  wo_to_bf16_kernel<<<(NDM * NDM + 255) / 256, 256, 0, stream>>>(Wo, Wob, NDM * NDM);
  qkv_proj_kernel<<<NB * NH * (NS / 64), 128, 0, stream>>>(
      q_in, k_in, v_in, Wq, bq, Wk, bk, Wv, bv, Qp, Kp, Vt);
  flash_attn_kernel<<<NB * NH * (NS / 64), 128, 0, stream>>>(Qp, Kp, Vt, At);
  out_proj_kernel<<<(NB * NS / 16) * (NDM / 256), 128, 0, stream>>>(At, Wob, bo, out);
}
